// WelkirSelfAttention_57664230916973
// MI455X (gfx1250) — matrix operand order repaired and checked by reading, not compiled, not run on hardware
//
#include <hip/hip_runtime.h>
#include <hip/hip_bf16.h>

// ---------------------------------------------------------------------------
// WelkirSelfAttention for MI455X (gfx1250): f16 WMMA everywhere.
//   B=4, S=1024, D=768, H=12, DH=64, MAXPOS=1024, NTYPES=8
// Pipeline:
//   1) f32->f16 conversion of X, Wq/Wk/Wv, dist_emb
//   2) QKV GEMM (WMMA 16x16x32 f16, f32 accum), 16x64 tile/wave, head-major out
//   3) Qp/Kp = {Q,K} x dist_emb^T ([49152x64]x[64x2048], 16x128 tile/wave)
//   4) Fused flash attention: QK^T (WMMA) + Qp/Kp gathers + flow-type bias,
//      online softmax, P*V (WMMA), f32 output [B,S,H*DH]
// ---------------------------------------------------------------------------

typedef __attribute__((ext_vector_type(16))) _Float16 v16h;
typedef __attribute__((ext_vector_type(8)))  _Float16 v8h;
typedef __attribute__((ext_vector_type(8)))  float    v8f;
typedef __attribute__((ext_vector_type(4)))  int      v4i;

constexpr int Bc  = 4;
constexpr int Sc  = 1024;
constexpr int Dc  = 768;
constexpr int Hc  = 12;
constexpr int DHc = 64;
constexpr int JW  = 2048;             // padded 2*MAXPOS-1 = 2047 -> 2048
constexpr size_t MQ = (size_t)Bc * Hc * Sc;   // 49152 rows of Q/K/V (head-major)

// Async global->LDS probe (gfx1250 GLOBAL_LOAD_ASYNC_TO_LDS_B128, ASYNCcnt).
#if defined(__AMDGCN__) && \
    __has_builtin(__builtin_amdgcn_global_load_async_to_lds_b128) && \
    __has_builtin(__builtin_amdgcn_s_wait_asynccnt)
#define USE_ASYNC_LDS 1
#else
#define USE_ASYNC_LDS 0
#endif

#if USE_ASYNC_LDS
// Exact pointer types per clang's diagnostic: v4i in addrspace(1)/(3).
__device__ __forceinline__ void async_g2l_b128(const void* g, void* l) {
    __builtin_amdgcn_global_load_async_to_lds_b128(
        (__attribute__((address_space(1))) v4i*)(void*)g,
        (__attribute__((address_space(3))) v4i*)l, 0, 0);
}
#endif

// A-fragment (16x32 f16): lane l -> M = l&15, hi = l>>4.
// a[0..7]  = row[k0 + 16*hi .. +7],  a[8..15] = row[k0 + 16*hi + 8 .. +7]
__device__ __forceinline__ v16h frag_a(const _Float16* p0, const _Float16* p1) {
    v8h lo = *(const v8h*)p0;
    v8h hi = *(const v8h*)p1;
    v16h r;
#pragma unroll
    for (int i = 0; i < 8; ++i) { r[i] = lo[i]; r[i + 8] = hi[i]; }
    return r;
}

// B-fragment (32x16 f16): lane l -> N = l&15, hi = l>>4.
// b[0..15] = col-N data at K = k0 + 16*hi .. +15 (K fastest in staged tile).
__device__ __forceinline__ v16h frag_b(const _Float16* p) {
    v8h lo = *(const v8h*)p;
    v8h hi = *(const v8h*)(p + 8);
    v16h r;
#pragma unroll
    for (int i = 0; i < 8; ++i) { r[i] = lo[i]; r[i + 8] = hi[i]; }
    return r;
}

__device__ __forceinline__ v8f wmma_f16(v16h a, v16h b, v8f c) {
    return __builtin_amdgcn_wmma_f32_16x16x32_f16(false, a, false, b,
                                                  (short)0, c, false, false);
}

// ---------------------------------------------------------------------------
// 1) f32 -> f16 convert
// ---------------------------------------------------------------------------
__global__ void cvt_f32_f16_kernel(const float* __restrict__ in,
                                   _Float16* __restrict__ out, int n) {
    int i = blockIdx.x * blockDim.x + threadIdx.x;
    if (i < n) out[i] = (_Float16)in[i];
}

// ---------------------------------------------------------------------------
// 2) QKV GEMM: Out[m,n] = sum_d X[m,d] * W[n,d] + bias[n]
//    Register-blocked: each wave computes a 16x64 tile (A reused over 4 B's).
//    grid = (64, 12, 3), block = 128 (4 waves)
// ---------------------------------------------------------------------------
__global__ __launch_bounds__(128) void qkv_gemm_kernel(
    const _Float16* __restrict__ Xh, const _Float16* __restrict__ WhAll,
    const float* __restrict__ bq, const float* __restrict__ bk,
    const float* __restrict__ bv,
    _Float16* __restrict__ Qh, _Float16* __restrict__ Kh,
    _Float16* __restrict__ Vh) {
    const int lane = threadIdx.x & 31;
    const int wave = threadIdx.x >> 5;
    const int ln = lane & 15, hi = lane >> 4;

    const int m0 = (blockIdx.x * 4 + wave) * 16;   // row tile in [0,4096)
    const int n0 = blockIdx.y * 64;                // 4 col tiles in [0,768)
    const int z  = blockIdx.z;                     // 0=Q 1=K 2=V

    const _Float16* W   = WhAll + (size_t)z * Dc * Dc;
    const float* bias   = (z == 0) ? bq : (z == 1) ? bk : bv;
    _Float16* Out       = (z == 0) ? Qh : (z == 1) ? Kh : Vh;

    const _Float16* arow = Xh + (size_t)(m0 + ln) * Dc;
    const _Float16* brow0 = W + (size_t)(n0 + 0  + ln) * Dc;  // B[k][n] = W[n][k]
    const _Float16* brow1 = W + (size_t)(n0 + 16 + ln) * Dc;
    const _Float16* brow2 = W + (size_t)(n0 + 32 + ln) * Dc;
    const _Float16* brow3 = W + (size_t)(n0 + 48 + ln) * Dc;

    v8f acc[4];
#pragma unroll
    for (int t = 0; t < 4; ++t) acc[t] = (v8f){0.f,0.f,0.f,0.f,0.f,0.f,0.f,0.f};

#pragma unroll 2
    for (int k0 = 0; k0 < Dc; k0 += 32) {
        v16h a = frag_a(arow + k0 + 16 * hi, arow + k0 + 8 + 16 * hi);
        acc[0] = wmma_f16(a, frag_b(brow0 + k0 + 16 * hi), acc[0]);
        acc[1] = wmma_f16(a, frag_b(brow1 + k0 + 16 * hi), acc[1]);
        acc[2] = wmma_f16(a, frag_b(brow2 + k0 + 16 * hi), acc[2]);
        acc[3] = wmma_f16(a, frag_b(brow3 + k0 + 16 * hi), acc[3]);
    }

    const int h = n0 >> 6;                 // n0 multiple of 64 -> single head
#pragma unroll
    for (int t = 0; t < 4; ++t) {
        const float bn = bias[n0 + t * 16 + ln];
        const int dh = t * 16 + ln;
#pragma unroll
        for (int v = 0; v < 8; ++v) {
            int m  = m0 + v + 8 * hi;      // m = b*S + s
            int bb = m >> 10, s = m & (Sc - 1);
            Out[((size_t)(bb * Hc + h) * Sc + s) * DHc + dh] =
                (_Float16)(acc[t][v] + bn);
        }
    }
}

// ---------------------------------------------------------------------------
// 3) Qp/Kp GEMM: P[m,j] = sum_d Src[m,d] * E[j,d]
//    Register-blocked: 16 rows x 8 j-tiles per wave; A frags resident.
//    grid = (768, 16, 2), block = 128
// ---------------------------------------------------------------------------
__global__ __launch_bounds__(128) void relpos_gemm_kernel(
    const _Float16* __restrict__ Qh, const _Float16* __restrict__ Kh,
    const _Float16* __restrict__ Eh,
    _Float16* __restrict__ Qp, _Float16* __restrict__ Kp) {
    const int lane = threadIdx.x & 31;
    const int wave = threadIdx.x >> 5;
    const int ln = lane & 15, hi = lane >> 4;

    const int m0 = (blockIdx.x * 4 + wave) * 16;   // row in [0,49152)
    const int j0 = blockIdx.y * 128;               // 8 distance tiles

    const _Float16* src = blockIdx.z ? Kh : Qh;
    _Float16* dst       = blockIdx.z ? Kp : Qp;

    const _Float16* arow = src + (size_t)(m0 + ln) * DHc;
    const v16h a0 = frag_a(arow + 0  + 16 * hi, arow + 8  + 16 * hi);
    const v16h a1 = frag_a(arow + 32 + 16 * hi, arow + 40 + 16 * hi);

    v8f acc[8];
#pragma unroll
    for (int t = 0; t < 8; ++t) acc[t] = (v8f){0.f,0.f,0.f,0.f,0.f,0.f,0.f,0.f};

#pragma unroll
    for (int t = 0; t < 8; ++t) {
        int jr = j0 + t * 16 + ln; if (jr > 2046) jr = 2046;  // clamp pad col
        const _Float16* brow = Eh + (size_t)jr * DHc;         // B[k=d][n=j]
        acc[t] = wmma_f16(a0, frag_b(brow + 0  + 16 * hi), acc[t]);
        acc[t] = wmma_f16(a1, frag_b(brow + 32 + 16 * hi), acc[t]);
    }

#pragma unroll
    for (int t = 0; t < 8; ++t)
#pragma unroll
        for (int v = 0; v < 8; ++v)
            dst[(size_t)(m0 + v + 8 * hi) * JW + j0 + t * 16 + ln] =
                (_Float16)acc[t][v];
}

// ---------------------------------------------------------------------------
// 4) Fused flash attention.
//    grid = (S/64, B*H), block = 128 (4 waves; wave w owns 16 query rows)
// ---------------------------------------------------------------------------
__global__ __launch_bounds__(128) void flash_attn_kernel(
    const _Float16* __restrict__ Qh, const _Float16* __restrict__ Kh,
    const _Float16* __restrict__ Vh,
    const _Float16* __restrict__ Qp, const _Float16* __restrict__ Kp,
    const float* __restrict__ cfg_emb, const float* __restrict__ dfg_emb,
    const float* __restrict__ rfg_emb,
    const int* __restrict__ cfgM, const int* __restrict__ dfgM,
    const int* __restrict__ rdgM,
    float* __restrict__ out) {
    __shared__ _Float16 sK[64 * 64];        // K tile, row-major [r][d]
    __shared__ _Float16 sVt[64 * 64];       // V tile, transposed [d][r]
    __shared__ _Float16 sP[4][16 * 64];     // per-wave probs tile [m][r]
    __shared__ float    sE[3][16];          // flow-type scalar tables

    const int tid  = threadIdx.x;
    const int lane = tid & 31, wave = tid >> 5;
    const int ln = lane & 15, hi = lane >> 4;

    const int bh = blockIdx.y;              // 0..47
    const int b  = bh / Hc;
    const int h  = bh % Hc;
    const int lw = blockIdx.x * 64 + wave * 16;   // this wave's first query row

    if (tid < 9) {
        sE[0][tid] = tid ? cfg_emb[tid] : 0.f;
        sE[1][tid] = tid ? dfg_emb[tid] : 0.f;
        sE[2][tid] = tid ? rfg_emb[tid] : 0.f;
    }

    const _Float16* Qbase = Qh + (size_t)bh * Sc * DHc;
    const _Float16* Kbase = Kh + (size_t)bh * Sc * DHc;
    const _Float16* Vbase = Vh + (size_t)bh * Sc * DHc;

    // Q fragments for this wave's 16 rows (resident for whole kernel).
    const _Float16* qrow = Qbase + (size_t)(lw + ln) * DHc;
    const v16h aQ0 = frag_a(qrow + 0  + 16 * hi, qrow + 8  + 16 * hi);
    const v16h aQ1 = frag_a(qrow + 32 + 16 * hi, qrow + 40 + 16 * hi);

    v8f O[4];
#pragma unroll
    for (int t = 0; t < 4; ++t) O[t] = (v8f){0.f,0.f,0.f,0.f,0.f,0.f,0.f,0.f};
    float rm[8], rl[8];
#pragma unroll
    for (int v = 0; v < 8; ++v) { rm[v] = -1e30f; rl[v] = 0.f; }

    const float inv_sqrt = 0.125f;          // 1/sqrt(64)

    for (int r0 = 0; r0 < Sc; r0 += 64) {
        __syncthreads();
        {   // Cooperative stage: K row-major (async if available), V transposed.
            const int row = tid >> 1;             // 0..63
            const int c0  = (tid & 1) * 32;       // 0 or 32 (halfs)
            const _Float16* ksrc = Kbase + (size_t)(r0 + row) * DHc + c0;
            _Float16* kdst = &sK[row * 64 + c0];
#if USE_ASYNC_LDS
#pragma unroll
            for (int q = 0; q < 4; ++q)
                async_g2l_b128(ksrc + q * 8, kdst + q * 8);
#else
#pragma unroll
            for (int q = 0; q < 4; ++q)
                *(uint4*)(kdst + q * 8) = *(const uint4*)(ksrc + q * 8);
#endif
            const _Float16* vsrc = Vbase + (size_t)(r0 + row) * DHc + c0;
#pragma unroll
            for (int q = 0; q < 4; ++q) {
                v8h vv = *(const v8h*)(vsrc + q * 8);
#pragma unroll
                for (int j = 0; j < 8; ++j)
                    sVt[(c0 + q * 8 + j) * 64 + row] = vv[j];
            }
            if (r0 + 64 < Sc) {   // speculative prefetch of next K/V tiles
                __builtin_prefetch(Kbase + (size_t)(r0 + 64 + row) * DHc + c0, 0, 0);
                __builtin_prefetch(Vbase + (size_t)(r0 + 64 + row) * DHc + c0, 0, 0);
            }
        }
#if USE_ASYNC_LDS
        __builtin_amdgcn_s_wait_asynccnt(0);
#endif
        __syncthreads();

        // ---- scores: 4 column tiles of 16, K-dim 64 -> 2 WMMAs each.
        //      aQ0 pass then aQ1 pass (A operand stays hot). ----
        v8f sc[4];
#pragma unroll
        for (int nt = 0; nt < 4; ++nt) sc[nt] = (v8f){0.f,0.f,0.f,0.f,0.f,0.f,0.f,0.f};
#pragma unroll
        for (int nt = 0; nt < 4; ++nt)
            sc[nt] = wmma_f16(aQ0, frag_b(&sK[(nt * 16 + ln) * 64 + 0  + 16 * hi]), sc[nt]);
#pragma unroll
        for (int nt = 0; nt < 4; ++nt)
            sc[nt] = wmma_f16(aQ1, frag_b(&sK[(nt * 16 + ln) * 64 + 32 + 16 * hi]), sc[nt]);

        // ---- biases: relative-position (Qp/Kp gathers) + flow-type ----
#pragma unroll
        for (int nt = 0; nt < 4; ++nt) {
            const int r = r0 + nt * 16 + ln;
#pragma unroll
            for (int v = 0; v < 8; ++v) {
                const int l = lw + v + 8 * hi;
                const int j = l - r + 1023;            // [0, 2046]
                float qb = (float)Qp[((size_t)bh * Sc + l) * JW + j];
                float kb = (float)Kp[((size_t)bh * Sc + r) * JW + j];
                size_t fi = ((size_t)b * Sc + l) * Sc + r;
                float fl = sE[0][cfgM[fi]] + sE[1][dfgM[fi]] + sE[2][rdgM[fi]];
                sc[nt][v] = (sc[nt][v] + qb + kb) * inv_sqrt + fl;
            }
        }

        // ---- online softmax update ----
#pragma unroll
        for (int v = 0; v < 8; ++v) {
            float m = sc[0][v];
#pragma unroll
            for (int nt = 1; nt < 4; ++nt) m = fmaxf(m, sc[nt][v]);
#pragma unroll
            for (int d = 1; d < 16; d <<= 1) m = fmaxf(m, __shfl_xor(m, d, 32));
            float nm = fmaxf(rm[v], m);
            float alpha = __expf(rm[v] - nm);
            rm[v] = nm;
            rl[v] *= alpha;
#pragma unroll
            for (int t = 0; t < 4; ++t) O[t][v] *= alpha;
        }

        // ---- P = exp(s - m): stage to LDS (row-major [m][r]) + row sums ----
#pragma unroll
        for (int nt = 0; nt < 4; ++nt) {
            float pr[8];
#pragma unroll
            for (int v = 0; v < 8; ++v) {
                float p = __expf(sc[nt][v] - rm[v]);
                pr[v] = p;
                sP[wave][(v + 8 * hi) * 64 + nt * 16 + ln] = (_Float16)p;
            }
#pragma unroll
            for (int v = 0; v < 8; ++v) {
                float ps = pr[v];
#pragma unroll
                for (int d = 1; d < 16; d <<= 1) ps += __shfl_xor(ps, d, 32);
                rl[v] += ps;
            }
        }
        // LDS ops from one wave are in-order, but be explicit before re-reading.
        asm volatile("s_wait_dscnt 0x0" ::: "memory");

        // ---- O += P * V  (A = probs 16x64 via LDS, B = V^T tiles) ----
        const _Float16* pb = &sP[wave][0];
        const v16h aP0 = frag_a(pb + ln * 64 + 0  + 16 * hi, pb + ln * 64 + 8  + 16 * hi);
        const v16h aP1 = frag_a(pb + ln * 64 + 32 + 16 * hi, pb + ln * 64 + 40 + 16 * hi);
#pragma unroll
        for (int dt = 0; dt < 4; ++dt)
            O[dt] = wmma_f16(aP0, frag_b(&sVt[(dt * 16 + ln) * 64 + 0  + 16 * hi]), O[dt]);
#pragma unroll
        for (int dt = 0; dt < 4; ++dt)
            O[dt] = wmma_f16(aP1, frag_b(&sVt[(dt * 16 + ln) * 64 + 32 + 16 * hi]), O[dt]);
    }

    // ---- finalize: out[b, l, h*64 + d] = O / rowsum ----
#pragma unroll
    for (int dt = 0; dt < 4; ++dt) {
#pragma unroll
        for (int v = 0; v < 8; ++v) {
            const int l = lw + v + 8 * hi;
            out[(((size_t)b * Sc + l) * Hc + h) * DHc + dt * 16 + ln] =
                O[dt][v] / rl[v];
        }
    }
}

// ---------------------------------------------------------------------------
// Host launcher
// ---------------------------------------------------------------------------
extern "C" void kernel_launch(void* const* d_in, const int* in_sizes, int n_in,
                              void* d_out, int out_size, void* d_ws, size_t ws_size,
                              hipStream_t stream) {
    const float* X    = (const float*)d_in[0];
    const float* Wq   = (const float*)d_in[1];
    const float* bq   = (const float*)d_in[2];
    const float* Wk   = (const float*)d_in[3];
    const float* bk   = (const float*)d_in[4];
    const float* Wv   = (const float*)d_in[5];
    const float* bv   = (const float*)d_in[6];
    const float* Emb  = (const float*)d_in[7];   // [2047, 64]
    const float* cfgE = (const float*)d_in[8];
    const float* dfgE = (const float*)d_in[9];
    const float* rfgE = (const float*)d_in[10];
    const int*   cfgM = (const int*)d_in[11];
    const int*   dfgM = (const int*)d_in[12];
    const int*   rdgM = (const int*)d_in[13];
    float* out = (float*)d_out;

    // Workspace carve-up (f16 buffers).  Total ~431 MB.
    char* ws = (char*)d_ws;
    size_t off = 0;
    auto carve = [&](size_t elems) {
        void* p = ws + off;
        off = (off + elems * sizeof(_Float16) + 255) & ~(size_t)255;
        return (_Float16*)p;
    };
    _Float16* Xh = carve((size_t)Bc * Sc * Dc);
    _Float16* Wh = carve((size_t)3 * Dc * Dc);     // Wq|Wk|Wv contiguous
    _Float16* Eh = carve((size_t)2047 * DHc);
    _Float16* Qh = carve(MQ * DHc);
    _Float16* Kh = carve(MQ * DHc);
    _Float16* Vh = carve(MQ * DHc);
    _Float16* Qp = carve(MQ * JW);
    _Float16* Kp = carve(MQ * JW);

    auto cvt = [&](const float* src, _Float16* dst, int n) {
        cvt_f32_f16_kernel<<<(n + 255) / 256, 256, 0, stream>>>(src, dst, n);
    };
    cvt(X,   Xh,                Bc * Sc * Dc);
    cvt(Wq,  Wh + 0 * (size_t)Dc * Dc, Dc * Dc);
    cvt(Wk,  Wh + 1 * (size_t)Dc * Dc, Dc * Dc);
    cvt(Wv,  Wh + 2 * (size_t)Dc * Dc, Dc * Dc);
    cvt(Emb, Eh,                2047 * DHc);

    qkv_gemm_kernel<<<dim3(64, 12, 3), 128, 0, stream>>>(
        Xh, Wh, bq, bk, bv, Qh, Kh, Vh);

    relpos_gemm_kernel<<<dim3(768, 16, 2), 128, 0, stream>>>(
        Qh, Kh, Eh, Qp, Kp);

    flash_attn_kernel<<<dim3(Sc / 64, Bc * Hc), 128, 0, stream>>>(
        Qh, Kh, Vh, Qp, Kp, cfgE, dfgE, rfgE, cfgM, dfgM, rdgM, out);
}
